// LocalMultiheadAttention_86947317941211
// MI455X (gfx1250) — compile-verified, matrix operand order checked by reading, not run on hardware
//
#include <hip/hip_runtime.h>

// ---------------------------------------------------------------------------
// LocalMultiheadAttention for MI455X (gfx1250), wave32 + WMMA bf16 + TDM.
//   T=2048, B=4, E=512, H=8, Dh=64.  scaling = 1/8.
// d_out = [ attn fp32 (T*B*E = 4,194,304) | avg_weights fp32 (B*T*T = 16,777,216) ]
// ---------------------------------------------------------------------------

#define T_LEN 2048
#define BSZ   4
#define EMB   512
#define NH    8
#define DH    64
#define ROWS  (T_LEN * BSZ)     // 8192
#define F3E   (3 * EMB)         // 1536

typedef __bf16 bf16;
typedef __attribute__((ext_vector_type(16))) __bf16 v16bf;
typedef __attribute__((ext_vector_type(8)))  float  v8f;
typedef __attribute__((ext_vector_type(4)))  unsigned int u32x4;
typedef __attribute__((ext_vector_type(8)))  int  i32x8;
typedef __attribute__((ext_vector_type(4)))  int  i32x4;

#if __has_builtin(__builtin_amdgcn_tensor_load_to_lds) && \
    __has_builtin(__builtin_amdgcn_s_wait_tensorcnt)
#define HAVE_TDM 1
#else
#define HAVE_TDM 0
#endif

// LDS float-offsets inside dynamic smem
#define OFF_SCORES 0
#define OFF_AVG    (16 * T_LEN)            // 32768
#define OFF_PART   (32 * T_LEN)            // 65536
#define OFF_RED    (OFF_PART + 8 * 256)    // 67584
#define OFF_ROWMAX (OFF_RED + 256)         // 67840
#define OFF_LINV   (OFF_ROWMAX + 16)       // 67856
#define OFF_VSTAGE (OFF_LINV + 16)         // 67872 floats = 271488 bytes
#define VSTAGE_BYTES  (64 * DH * 2)        // one 64x64 bf16 chunk = 8192 B
#define LDS_TOTAL_BYTES (OFF_VSTAGE * 4 + 2 * VSTAGE_BYTES)  // 287872 < 320K

__device__ __forceinline__ v8f wmma_bf16(v16bf a, v16bf b, v8f c) {
  return __builtin_amdgcn_wmma_f32_16x16x32_bf16(false, a, false, b, (short)0, c,
                                                 false, false);
}

// K-index of element pair base for VGPR v (ISA 16-bit A layout).
__device__ __forceinline__ int frag_kbase(int v, int half) {
  return (v < 4) ? (2 * v + 8 * half) : (16 + 2 * (v - 4) + 8 * half);
}

// A-style fragment: 16 rows (lane&15) x 32 K, row-major source, pairs contiguous.
__device__ __forceinline__ v16bf ld_frag(const bf16* __restrict__ p, int ld, int lane) {
  int row = lane & 15, half = lane >> 4;
  const bf16* r = p + (size_t)row * ld;
  v16bf f;
#pragma unroll
  for (int v = 0; v < 8; ++v) {
    int kb = frag_kbase(v, half);
    f[2 * v]     = r[kb];
    f[2 * v + 1] = r[kb + 1];
  }
  return f;
}

// B operand from [K][N]-stored source (V chunk in LDS): element [n][k] = p[k*ld+col].
__device__ __forceinline__ v16bf ld_fragT(const bf16* p, int ld, int col, int lane) {
  int half = lane >> 4;
  v16bf f;
#pragma unroll
  for (int v = 0; v < 8; ++v) {
    int kb = frag_kbase(v, half);
    f[2 * v]     = p[kb * ld + col];
    f[2 * v + 1] = p[(kb + 1) * ld + col];
  }
  return f;
}

// P fragment from fp32 LDS scores, scaled (1/l) and converted to bf16 on the fly.
__device__ __forceinline__ v16bf ld_fragP(const float* sc, int ld, int k0, float scale,
                                          int lane) {
  int row = lane & 15, half = lane >> 4;
  const float* r = sc + row * ld + k0;
  v16bf f;
#pragma unroll
  for (int v = 0; v < 8; ++v) {
    int kb = frag_kbase(v, half);
    f[2 * v]     = (bf16)(r[kb] * scale);
    f[2 * v + 1] = (bf16)(r[kb + 1] * scale);
  }
  return f;
}

#if HAVE_TDM
// Issue a TDM load of one 64-row x 64-col bf16 tile of the V slab into LDS.
// D# per ISA ch.8: tensor = [2048][64] bf16, stride0 = 64; tile = 64x64;
// global_addr = tile start; lds_addr = byte offset in workgroup LDS.
__device__ __forceinline__ void tdm_load_v_chunk(const bf16* gsrc, unsigned lds_off) {
  unsigned long long ga = (unsigned long long)(__UINTPTR_TYPE__)gsrc;
  u32x4 g0;
  g0[0] = 1u;                                             // count=1 (valid user D#)
  g0[1] = lds_off;                                        // lds_addr
  g0[2] = (unsigned)(ga & 0xffffffffu);                   // global_addr[31:0]
  g0[3] = (unsigned)((ga >> 32) & 0x01ffffffu) | (2u << 30);  // addr[56:32]|type=2
  i32x8 g1;
  g1[0] = (1 << 16);          // workgroup_mask=0, data_size=1 (2 bytes), no flags
  g1[1] = (64 << 16);         // atomic_barrier_addr=0 | tensor_dim0 = 64
  g1[2] = (2048 << 16);       // tensor_dim0 hi=0 | tensor_dim1 = 2048
  g1[3] = (64 << 16);         // tensor_dim1 hi=0 | tile_dim0 = 64
  g1[4] = 64;                 // tile_dim1 = 64 rows, tile_dim2 = 0
  g1[5] = 64;                 // tensor_dim0_stride = 64
  g1[6] = 0;                  // stride0 hi | tensor_dim1_stride lo (unused, 2D tile)
  g1[7] = 0;
  i32x4 z4 = {0, 0, 0, 0};
#if __clang_major__ >= 23
  i32x8 z8 = {0, 0, 0, 0, 0, 0, 0, 0};
  __builtin_amdgcn_tensor_load_to_lds(g0, g1, z4, z4, z8, 0);
#else
  __builtin_amdgcn_tensor_load_to_lds(g0, g1, z4, z4, 0);
#endif
}
#endif

// ---------------------------------------------------------------------------
// Kernel 0: fp32 -> bf16 conversions (query, in_proj_weight, out_w)
// ---------------------------------------------------------------------------
__global__ void cvt_kernel(const float* __restrict__ q, const float* __restrict__ wi,
                           const float* __restrict__ wo, bf16* __restrict__ qb,
                           bf16* __restrict__ wib, bf16* __restrict__ wob) {
  size_t i = (size_t)blockIdx.x * blockDim.x + threadIdx.x;
  if (i < (size_t)ROWS * EMB) qb[i] = (bf16)q[i];
  if (i < (size_t)F3E * EMB)  wib[i] = (bf16)wi[i];
  if (i < (size_t)EMB * EMB)  wob[i] = (bf16)wo[i];
}

// ---------------------------------------------------------------------------
// Kernel 1: QKV projection (M=8192, N=1536, K=512), one 16x16 tile per wave.
// ---------------------------------------------------------------------------
__global__ void qkv_gemm_kernel(const bf16* __restrict__ A, const bf16* __restrict__ W,
                                const float* __restrict__ bias, bf16* __restrict__ qkv) {
  int wave = threadIdx.x >> 5, lane = threadIdx.x & 31;
  int tile = blockIdx.x * 8 + wave;
  int tn = tile % (F3E / 16);
  int tm = tile / (F3E / 16);
  const bf16* Arow = A + (size_t)tm * 16 * EMB;
  const bf16* Wrow = W + (size_t)tn * 16 * EMB;
  v8f c = {};
#pragma unroll 4
  for (int k0 = 0; k0 < EMB; k0 += 32) {
    __builtin_prefetch(Arow + k0 + 256, 0, 1);
    v16bf a = ld_frag(Arow + k0, EMB, lane);
    v16bf b = ld_frag(Wrow + k0, EMB, lane);
    c = wmma_bf16(a, b, c);
  }
  int col = lane & 15, half = lane >> 4;
  int f = tn * 16 + col;
  float bv = bias[f];
  int which = f >> 9;
  int rr = f & 511;
  int h = rr >> 6, d = rr & 63;
  float scale = (which == 0) ? 0.125f : 1.0f;
#pragma unroll
  for (int v = 0; v < 8; ++v) {
    int mr = v + 8 * half;
    int grow = tm * 16 + mr;                 // row = t*B + b
    int t = grow >> 2, bb = grow & 3;
    float val = (c[v] + bv) * scale;
    qkv[(((size_t)which * BSZ + bb) * NH + h) * T_LEN * DH + (size_t)t * DH + d] =
        (bf16)val;
  }
}

// ---------------------------------------------------------------------------
// Kernel 2: attention core. Block = (b, 16-row Q tile); loops over 8 heads.
// V is staged into LDS in 64x64 chunks by the Tensor Data Mover (double buffer).
// ---------------------------------------------------------------------------
__global__ void attn_kernel(const bf16* __restrict__ qkv, bf16* __restrict__ attn_bf,
                            float* __restrict__ avg_out) {
  extern __shared__ float smem[];
  float* scores = smem + OFF_SCORES;            // [16][2048]
  float* avg    = smem + OFF_AVG;               // [16][2048]
  float* part   = smem + OFF_PART;              // [8][16][16]
  float* red    = smem + OFF_RED;               // [16][16]
  float* rowmax = smem + OFF_ROWMAX;            // [16]
  float* linv   = smem + OFF_LINV;              // [16]
  bf16*  vstage = (bf16*)(smem + OFF_VSTAGE);   // 2 x [64][64] bf16

  const int tid  = threadIdx.x;
  const int wave = tid >> 5, lane = tid & 31;
  const int col  = lane & 15, half = lane >> 4;
  const int t0   = blockIdx.x * 16;
  const int b    = blockIdx.y;

  for (int i = tid; i < 16 * T_LEN; i += 256) avg[i] = 0.0f;
  __syncthreads();

  for (int h = 0; h < NH; ++h) {
    const bf16* Qb = qkv + (((size_t)0 * BSZ + b) * NH + h) * T_LEN * DH;
    const bf16* Kb = qkv + (((size_t)1 * BSZ + b) * NH + h) * T_LEN * DH;
    const bf16* Vb = qkv + (((size_t)2 * BSZ + b) * NH + h) * T_LEN * DH;

    // ---- scores = Q Kᵀ (Q pre-scaled), minus max(0, log|i-j|) ----
    v16bf qa0 = ld_frag(Qb + (size_t)t0 * DH, DH, lane);
    v16bf qa1 = ld_frag(Qb + (size_t)t0 * DH + 32, DH, lane);
    for (int st = wave; st < T_LEN / 16; st += 8) {
      int s0 = st * 16;
      v16bf kb0 = ld_frag(Kb + (size_t)s0 * DH, DH, lane);
      v16bf kb1 = ld_frag(Kb + (size_t)s0 * DH + 32, DH, lane);
      v8f c = {};
      c = wmma_bf16(qa0, kb0, c);
      c = wmma_bf16(qa1, kb1, c);
#pragma unroll
      for (int v = 0; v < 8; ++v) {
        int mr = v + 8 * half;
        float diff = fabsf((float)(t0 + mr) - (float)(s0 + col));
        float pen  = fmaxf(0.0f, __logf(diff));   // log(0)->-inf -> 0
        scores[mr * T_LEN + s0 + col] = c[v] - pen;
      }
    }
    __syncthreads();

    // ---- two-pass softmax ----
    int r = tid & 15, g = tid >> 4;
    float* srow = scores + r * T_LEN + g * 128;
    float mx = -1e30f;
    for (int s = 0; s < 128; ++s) mx = fmaxf(mx, srow[s]);
    red[r * 16 + g] = mx;
    __syncthreads();
    if (tid < 16) {
      float m = -1e30f;
      for (int gg = 0; gg < 16; ++gg) m = fmaxf(m, red[tid * 16 + gg]);
      rowmax[tid] = m;
    }
    __syncthreads();
    float m = rowmax[r];
    float sum = 0.0f;
    for (int s = 0; s < 128; ++s) {
      float e = __expf(srow[s] - m);
      srow[s] = e;
      sum += e;
    }
    red[r * 16 + g] = sum;
    __syncthreads();
    if (tid < 16) {
      float l = 0.0f;
      for (int gg = 0; gg < 16; ++gg) l += red[tid * 16 + gg];
      linv[tid] = 1.0f / l;
    }
    __syncthreads();

    // ---- accumulate head-averaged probs ----
    float rl = linv[r];
    float* arow = avg + r * T_LEN + g * 128;
    for (int s = 0; s < 128; ++s) arow[s] += srow[s] * rl * (1.0f / NH);

    // ---- attn tile = P V, V chunks (64 rows) staged into LDS ----
    // 8 waves: nt = wave&3 (N-tile of Dh), ks = wave>>2 (which 32-K half of chunk)
    int nt = wave & 3, ks = wave >> 2;
    int d  = nt * 16 + col;
    float rlrow = linv[lane & 15];
    v8f c = {};

#if HAVE_TDM
    if (wave == 0) tdm_load_v_chunk(Vb, OFF_VSTAGE * 4);   // prime buffer 0
#endif
    for (int kc = 0; kc < T_LEN / 64; ++kc) {              // 32 chunks
      int nbuf = (kc + 1) & 1;
#if HAVE_TDM
      if (wave == 0) {
        if (kc + 1 < T_LEN / 64) {
          tdm_load_v_chunk(Vb + (size_t)(kc + 1) * 64 * DH,
                           OFF_VSTAGE * 4 + nbuf * VSTAGE_BYTES);
          __builtin_amdgcn_s_wait_tensorcnt(1);            // chunk kc complete
        } else {
          __builtin_amdgcn_s_wait_tensorcnt(0);
        }
      }
#else
      // fallback: cooperative global->LDS copy of chunk kc (buffer kc&1)
      {
        const uint4* src = (const uint4*)(Vb + (size_t)kc * 64 * DH);
        uint4* dst = (uint4*)(vstage + (size_t)(kc & 1) * 64 * DH);
        for (int i = tid; i < (64 * DH * 2) / 16; i += 256) dst[i] = src[i];
      }
      (void)nbuf;
#endif
      __syncthreads();
      const bf16* vch = vstage + (size_t)(kc & 1) * 64 * DH;  // [64][64]
      v16bf pa = ld_fragP(scores, T_LEN, kc * 64 + ks * 32, rlrow, lane);
      v16bf vb = ld_fragT(vch + ks * 32 * DH, DH, d, lane);
      c = wmma_bf16(pa, vb, c);
      __syncthreads();
    }

#pragma unroll
    for (int v = 0; v < 8; ++v)
      part[wave * 256 + (v + 8 * half) * 16 + col] = c[v];
    __syncthreads();

    // combine K-halves and store bf16 tile into [t][b][h*64+d]
    for (int idx = tid; idx < 16 * DH; idx += 256) {
      int mr = idx >> 6, dd = idx & 63;
      int ntl = dd >> 4, nn = dd & 15;
      float vsum = part[ntl * 256 + mr * 16 + nn] + part[(ntl + 4) * 256 + mr * 16 + nn];
      attn_bf[(size_t)(t0 + mr) * (BSZ * EMB) + (size_t)b * EMB + h * DH + dd] =
          (bf16)vsum;
    }
    __syncthreads();
  }

  // ---- write avg weights tile [b][t0..t0+15][0..T) ----
  float* ao = avg_out + (size_t)b * T_LEN * T_LEN + (size_t)t0 * T_LEN;
  for (int idx = tid; idx < 16 * T_LEN; idx += 256) {
    int rr = idx >> 11, ss = idx & (T_LEN - 1);
    ao[(size_t)rr * T_LEN + ss] = avg[rr * T_LEN + ss];
  }
}

// ---------------------------------------------------------------------------
// Kernel 3: out projection (M=8192, N=512, K=512) -> fp32 d_out [T][B][E].
// ---------------------------------------------------------------------------
__global__ void out_gemm_kernel(const bf16* __restrict__ A, const bf16* __restrict__ W,
                                const float* __restrict__ bias, float* __restrict__ out) {
  int wave = threadIdx.x >> 5, lane = threadIdx.x & 31;
  int tile = blockIdx.x * 8 + wave;
  int tn = tile & 31;
  int tm = tile >> 5;
  const bf16* Arow = A + (size_t)tm * 16 * EMB;
  const bf16* Wrow = W + (size_t)tn * 16 * EMB;
  v8f c = {};
#pragma unroll 4
  for (int k0 = 0; k0 < EMB; k0 += 32) {
    __builtin_prefetch(Wrow + k0 + 256, 0, 1);
    v16bf a = ld_frag(Arow + k0, EMB, lane);
    v16bf b = ld_frag(Wrow + k0, EMB, lane);
    c = wmma_bf16(a, b, c);
  }
  int col = lane & 15, half = lane >> 4;
  int f = tn * 16 + col;
  float bv = bias[f];
#pragma unroll
  for (int v = 0; v < 8; ++v) {
    int mr = v + 8 * half;
    int grow = tm * 16 + mr;
    out[(size_t)grow * EMB + f] = c[v] + bv;
  }
}

// ---------------------------------------------------------------------------
extern "C" void kernel_launch(void* const* d_in, const int* in_sizes, int n_in,
                              void* d_out, int out_size, void* d_ws, size_t ws_size,
                              hipStream_t stream) {
  const float* query = (const float*)d_in[0];   // [T,B,E]
  const float* w_in  = (const float*)d_in[1];   // [3E,E]
  const float* b_in  = (const float*)d_in[2];   // [3E]
  const float* w_out = (const float*)d_in[3];   // [E,E]
  const float* b_out = (const float*)d_in[4];   // [E]
  float* out = (float*)d_out;
  float* avg_out = out + (size_t)ROWS * EMB;

  char* ws = (char*)d_ws;
  bf16* qb   = (bf16*)(ws);                                         // 8192*512
  bf16* wib  = (bf16*)(ws + (size_t)ROWS * EMB * 2);                // 1536*512
  bf16* wob  = (bf16*)(ws + (size_t)(ROWS + F3E) * EMB * 2);        // 512*512
  bf16* qkvb = (bf16*)(ws + (size_t)(ROWS + F3E + EMB) * EMB * 2);  // 3*B*H*T*64
  bf16* attb = (bf16*)(ws + (size_t)(ROWS + F3E + EMB) * EMB * 2 +
                       (size_t)3 * ROWS * EMB * 2);                 // 8192*512

  cvt_kernel<<<(ROWS * EMB) / 256, 256, 0, stream>>>(query, w_in, w_out, qb, wib, wob);
  qkv_gemm_kernel<<<(512 * 96) / 8, 256, 0, stream>>>(qb, wib, b_in, qkvb);
  attn_kernel<<<dim3(T_LEN / 16, BSZ), 256, (size_t)LDS_TOTAL_BYTES, stream>>>(
      qkvb, attb, avg_out);
  out_gemm_kernel<<<(512 * 32) / 8, 256, 0, stream>>>(attb, wob, b_out, out);
}
